// AttentionDecoder_63617055588602
// MI455X (gfx1250) — compile-verified, matrix operand order checked by reading, not run on hardware
//
#include <hip/hip_runtime.h>
#include <hip/hip_bf16.h>
#include <math.h>

// ---------------- problem constants ----------------
#define C_V    10000
#define C_E    256
#define C_H    512
#define C_ENC  512
#define C_ATT  512
#define C_B    128
#define C_S    25
#define C_NF   196
#define C_XK   1280   // E + ENC + H  (concat LSTM input)
#define C_G    2048   // 4*H

typedef __attribute__((ext_vector_type(16))) _Float16 v16h;
typedef __attribute__((ext_vector_type(8)))  float    v8f;
typedef __attribute__((ext_vector_type(4)))  float    f32x4;

// pack 16 f32 (two contiguous 8-float runs) into the v16h WMMA operand layout
__device__ __forceinline__ v16h pack16(f32x4 a0, f32x4 a1, f32x4 a2, f32x4 a3) {
    v16h v;
    v[0]  = (_Float16)a0[0]; v[1]  = (_Float16)a0[1];
    v[2]  = (_Float16)a0[2]; v[3]  = (_Float16)a0[3];
    v[4]  = (_Float16)a1[0]; v[5]  = (_Float16)a1[1];
    v[6]  = (_Float16)a1[2]; v[7]  = (_Float16)a1[3];
    v[8]  = (_Float16)a2[0]; v[9]  = (_Float16)a2[1];
    v[10] = (_Float16)a2[2]; v[11] = (_Float16)a2[3];
    v[12] = (_Float16)a3[0]; v[13] = (_Float16)a3[1];
    v[14] = (_Float16)a3[2]; v[15] = (_Float16)a3[3];
    return v;
}

// load one 16x32 (or K-row-of-W) f16 fragment for this lane: two contiguous
// 8-float runs at +0 and +16 (per CDNA5 16-bit A/B operand layout)
__device__ __forceinline__ v16h loadfrag(const float* __restrict__ p) {
    f32x4 x0 = *(const f32x4*)(p);
    f32x4 x1 = *(const f32x4*)(p + 4);
    f32x4 x2 = *(const f32x4*)(p + 16);
    f32x4 x3 = *(const f32x4*)(p + 20);
    return pack16(x0, x1, x2, x3);
}

#define WMMA_F16(Av, Bv, Cv) \
    __builtin_amdgcn_wmma_f32_16x16x32_f16(false, (Av), false, (Bv), (short)0, (Cv), false, false)

__device__ __forceinline__ float dsigmoid(float x) { return 1.0f / (1.0f + expf(-x)); }

// ---------------------------------------------------------------------------
// C[M,N] = A[M,K] (f32, row-major, lda) * W[N,K]^T (f32, row-major, ldw) + bias
// f16 WMMA 16x16x32, f32 accumulate. 256 thr = 8 waves arranged 2(M) x 4(N);
// each wave register-blocks a 32x32 patch (2x2 tiles, 4 WMMAs / k-step).
// Requires M%64==0, K%32==0, N%16==0. Ragged N handled with wave-uniform
// branches (EXEC stays all-ones for WMMA).
// ---------------------------------------------------------------------------
__global__ void __launch_bounds__(256)
gemm_f16_wmma(const float* __restrict__ A, int lda,
              const float* __restrict__ W, int ldw,
              const float* __restrict__ bias,
              float* __restrict__ C, long long ldc,
              int M, int N, int K)
{
    const int tid  = threadIdx.x;
    const int wave = tid >> 5;
    const int lane = tid & 31;
    const int hf   = lane >> 4;      // 0: K 0..7 / 16..23,  1: K 8..15 / 24..31
    const int r    = lane & 15;

    const int tilesN = N >> 4;
    const int nt0 = blockIdx.x * 8 + (wave & 3) * 2;
    const int mt0 = blockIdx.y * 4 + (wave >> 2) * 2;
    if (nt0 >= tilesN) return;                       // wave-uniform
    const bool hasN1 = (nt0 + 1) < tilesN;

    const int row0 = mt0 * 16;
    const int col0 = nt0 * 16;

    const float* Ar0 = A + (size_t)(row0 + r) * lda;
    const float* Ar1 = Ar0 + (size_t)16 * lda;
    const float* Wr0 = W + (size_t)(col0 + r) * ldw;
    const float* Wr1 = Wr0 + (size_t)16 * ldw;

    v8f acc00 = {}, acc01 = {}, acc10 = {}, acc11 = {};

    if (hasN1) {
        for (int k0 = 0; k0 < K; k0 += 32) {
            const int ka = k0 + hf * 8;
            __builtin_prefetch(Wr0 + ka + 128, 0, 1);   // ~4 k-steps ahead
            __builtin_prefetch(Wr1 + ka + 128, 0, 1);
            v16h a0 = loadfrag(Ar0 + ka);
            v16h a1 = loadfrag(Ar1 + ka);
            v16h b0 = loadfrag(Wr0 + ka);
            v16h b1 = loadfrag(Wr1 + ka);
            acc00 = WMMA_F16(a0, b0, acc00);
            acc01 = WMMA_F16(a0, b1, acc01);
            acc10 = WMMA_F16(a1, b0, acc10);
            acc11 = WMMA_F16(a1, b1, acc11);
        }
    } else {
#pragma unroll 2
        for (int k0 = 0; k0 < K; k0 += 32) {
            const int ka = k0 + hf * 8;
            __builtin_prefetch(Wr0 + ka + 128, 0, 1);
            v16h a0 = loadfrag(Ar0 + ka);
            v16h a1 = loadfrag(Ar1 + ka);
            v16h b0 = loadfrag(Wr0 + ka);
            acc00 = WMMA_F16(a0, b0, acc00);
            acc10 = WMMA_F16(a1, b0, acc10);
        }
    }

    // C/D layout: VGPR e -> row e (lanes 0-15) / row e+8 (lanes 16-31), col=r
    const float b0add = bias ? bias[col0 + r] : 0.0f;
    float* Cp00 = C + (size_t)(row0 + hf * 8) * ldc + col0 + r;
    float* Cp10 = Cp00 + (size_t)16 * ldc;
#pragma unroll
    for (int e = 0; e < 8; ++e) {
        Cp00[(size_t)e * ldc] = acc00[e] + b0add;
        Cp10[(size_t)e * ldc] = acc10[e] + b0add;
    }
    if (hasN1) {
        const float b1add = bias ? bias[col0 + 16 + r] : 0.0f;
        float* Cp01 = Cp00 + 16;
        float* Cp11 = Cp10 + 16;
#pragma unroll
        for (int e = 0; e < 8; ++e) {
            Cp01[(size_t)e * ldc] = acc01[e] + b1add;
            Cp11[(size_t)e * ldc] = acc11[e] + b1add;
        }
    }
}

// ---------------------------------------------------------------------------
// one-time: Wcat[2048][1280] = [W_ih | W_hh], bcat = b_ih + b_hh
// ---------------------------------------------------------------------------
__global__ void build_cat_kernel(const float* __restrict__ W_ih,
                                 const float* __restrict__ W_hh,
                                 const float* __restrict__ b_ih,
                                 const float* __restrict__ b_hh,
                                 float* __restrict__ Wcat,
                                 float* __restrict__ bcat)
{
    const size_t total  = (size_t)C_G * C_XK;
    const size_t stride = (size_t)gridDim.x * blockDim.x;
    size_t gid = (size_t)blockIdx.x * blockDim.x + threadIdx.x;
    for (size_t i = gid; i < total; i += stride) {
        const size_t row = i / C_XK, col = i % C_XK;
        Wcat[i] = (col < (C_E + C_ENC)) ? W_ih[row * (C_E + C_ENC) + col]
                                        : W_hh[row * C_H + (col - (C_E + C_ENC))];
    }
    if (gid < C_G) bcat[gid] = b_ih[gid] + b_hh[gid];
}

// mean over NF features: mean_enc[b][e] = avg_n enc[b][n][e]
__global__ void mean_kernel(const float* __restrict__ enc, float* __restrict__ mean_enc)
{
    const int b = blockIdx.x;
    const float* eb = enc + (size_t)b * C_NF * C_ENC;
    for (int e = threadIdx.x; e < C_ENC; e += blockDim.x) {
        float s = 0.0f;
        for (int n = 0; n < C_NF; ++n) s += eb[(size_t)n * C_ENC + e];
        mean_enc[(size_t)b * C_ENC + e] = s * (1.0f / C_NF);
    }
}

// copy h into the h-slot of xh (used once for h0)
__global__ void h2xh_kernel(const float* __restrict__ h, float* __restrict__ xh)
{
    const int idx = blockIdx.x * blockDim.x + threadIdx.x;   // 0 .. B*H-1
    const int b = idx >> 9, j = idx & (C_H - 1);
    xh[(size_t)b * C_XK + C_E + C_ENC + j] = h[idx];
}

// embeds -> xh[:, 0:E] ; blockDim == E
__global__ void embed_kernel(const int* __restrict__ dec,
                             const float* __restrict__ emb,
                             const float* __restrict__ pos_emb,
                             float* __restrict__ xh, int t)
{
    const int b = blockIdx.x, e = threadIdx.x;
    const int tok = dec[b * C_S + t];
    xh[(size_t)b * C_XK + e] = emb[(size_t)tok * C_E + e] + pos_emb[(size_t)t * C_E + e];
}

// ---------------------------------------------------------------------------
// attention: scores = tanh(u_hs + w_ah)·Aw + Ab ; softmax ; context -> xh slot
// one block per batch row, 8 waves; wave-per-score-row with lane reduction.
// ---------------------------------------------------------------------------
__global__ void __launch_bounds__(256)
attn_kernel(const float* __restrict__ u_hs, const float* __restrict__ wah,
            const float* __restrict__ enc,  const float* __restrict__ Aw,
            const float* __restrict__ Ab,   float* __restrict__ alphas_out,
            float* __restrict__ xh, int t)
{
    __shared__ float s_wah[C_ATT];
    __shared__ float s_sc[C_NF];
    __shared__ float s_red[8];

    const int b = blockIdx.x, tid = threadIdx.x;
    const int wave = tid >> 5, lane = tid & 31;

    for (int a = tid; a < C_ATT; a += 256) s_wah[a] = wah[(size_t)b * C_ATT + a];
    __syncthreads();

    const float* ub = u_hs + (size_t)b * C_NF * C_ATT;
    for (int n = wave; n < C_NF; n += 8) {
        const float* u = ub + (size_t)n * C_ATT;
        float p = 0.0f;
        for (int a = lane; a < C_ATT; a += 32)
            p += tanhf(u[a] + s_wah[a]) * Aw[a];
        for (int m = 16; m >= 1; m >>= 1) p += __shfl_xor(p, m, 32);
        if (lane == 0) s_sc[n] = p + Ab[0];
    }
    __syncthreads();

    // block softmax over NF=196
    float mx = -3.402823466e38f;
    for (int n = tid; n < C_NF; n += 256) mx = fmaxf(mx, s_sc[n]);
    for (int m = 16; m >= 1; m >>= 1) mx = fmaxf(mx, __shfl_xor(mx, m, 32));
    if (lane == 0) s_red[wave] = mx;
    __syncthreads();
    mx = s_red[0];
#pragma unroll
    for (int w = 1; w < 8; ++w) mx = fmaxf(mx, s_red[w]);

    float sm = 0.0f;
    for (int n = tid; n < C_NF; n += 256) {
        float e = expf(s_sc[n] - mx);
        s_sc[n] = e;
        sm += e;
    }
    for (int m = 16; m >= 1; m >>= 1) sm += __shfl_xor(sm, m, 32);
    __syncthreads();
    if (lane == 0) s_red[wave] = sm;
    __syncthreads();
    sm = 0.0f;
#pragma unroll
    for (int w = 0; w < 8; ++w) sm += s_red[w];
    const float inv = 1.0f / sm;

    for (int n = tid; n < C_NF; n += 256) {
        const float al = s_sc[n] * inv;
        s_sc[n] = al;
        alphas_out[((size_t)b * C_S + t) * C_NF + n] = al;
    }
    __syncthreads();

    // context[e] = sum_n alpha[n] * enc[b][n][e]  -> xh[:, E : E+ENC]
    const float* eb = enc + (size_t)b * C_NF * C_ENC;
    for (int e = tid; e < C_ENC; e += 256) {
        float ctx = 0.0f;
        for (int n = 0; n < C_NF; ++n) ctx += s_sc[n] * eb[(size_t)n * C_ENC + e];
        xh[(size_t)b * C_XK + C_E + e] = ctx;
    }
}

// LSTM pointwise; also feeds h_new into xh's h-slot for the next step
__global__ void lstm_kernel(const float* __restrict__ gates,
                            float* __restrict__ cbuf, float* __restrict__ hbuf,
                            float* __restrict__ xh)
{
    const int idx = blockIdx.x * blockDim.x + threadIdx.x;   // 0 .. B*H-1
    const int b = idx >> 9, j = idx & (C_H - 1);
    const float* g = gates + (size_t)b * C_G;
    const float ig = dsigmoid(g[j]);
    const float fg = dsigmoid(g[C_H + j]);
    const float gg = tanhf(g[2 * C_H + j]);
    const float og = dsigmoid(g[3 * C_H + j]);
    const float c  = fg * cbuf[idx] + ig * gg;
    const float h  = og * tanhf(c);
    cbuf[idx] = c;
    hbuf[idx] = h;
    xh[(size_t)b * C_XK + C_E + C_ENC + j] = h;
}

// ---------------------------------------------------------------------------
extern "C" void kernel_launch(void* const* d_in, const int* in_sizes, int n_in,
                              void* d_out, int out_size, void* d_ws, size_t ws_size,
                              hipStream_t stream)
{
    (void)in_sizes; (void)n_in; (void)out_size; (void)ws_size;

    const int*   dec     = (const int*)  d_in[0];
    const float* enc     = (const float*)d_in[1];
    const float* emb     = (const float*)d_in[2];
    const float* pos_emb = (const float*)d_in[3];
    const float* Uw      = (const float*)d_in[4];
    const float* Ub      = (const float*)d_in[5];
    const float* Ww      = (const float*)d_in[6];
    const float* Wb      = (const float*)d_in[7];
    const float* Aw      = (const float*)d_in[8];
    const float* Ab      = (const float*)d_in[9];
    const float* ihw     = (const float*)d_in[10];
    const float* ihb     = (const float*)d_in[11];
    const float* icw     = (const float*)d_in[12];
    const float* icb     = (const float*)d_in[13];
    const float* W_ih    = (const float*)d_in[14];
    const float* W_hh    = (const float*)d_in[15];
    const float* b_ih    = (const float*)d_in[16];
    const float* b_hh    = (const float*)d_in[17];
    const float* fcn_w   = (const float*)d_in[18];
    const float* fcn_b   = (const float*)d_in[19];

    float* out    = (float*)d_out;                           // [B,S,V]
    float* alphas = out + (size_t)C_B * C_S * C_V;           // [B,S,NF]

    // workspace carve-up (floats)
    float* ws = (float*)d_ws;
    size_t o = 0;
    float* u_hs  = ws + o; o += (size_t)C_B * C_NF * C_ATT;  // 12.8M
    float* Wcat  = ws + o; o += (size_t)C_G * C_XK;          // 2.6M
    float* bcat  = ws + o; o += C_G;
    float* meanb = ws + o; o += (size_t)C_B * C_ENC;
    float* hbuf  = ws + o; o += (size_t)C_B * C_H;
    float* cbuf  = ws + o; o += (size_t)C_B * C_H;
    float* wah   = ws + o; o += (size_t)C_B * C_ATT;
    float* xh    = ws + o; o += (size_t)C_B * C_XK;
    float* gates = ws + o; o += (size_t)C_B * C_G;

    auto gemm = [&](const float* A, int lda, const float* W, int ldw,
                    const float* bias, float* Cp, long long ldc,
                    int M, int N, int K) {
        dim3 grid((N / 16 + 7) / 8, M / 64);
        gemm_f16_wmma<<<grid, 256, 0, stream>>>(A, lda, W, ldw, bias, Cp, ldc, M, N, K);
    };

    // ---- loop-invariant precompute ----
    build_cat_kernel<<<1024, 256, 0, stream>>>(W_ih, W_hh, b_ih, b_hh, Wcat, bcat);
    gemm(enc, C_ENC, Uw, C_ENC, Ub, u_hs, C_ATT, C_B * C_NF, C_ATT, C_ENC);
    mean_kernel<<<C_B, 256, 0, stream>>>(enc, meanb);
    gemm(meanb, C_ENC, ihw, C_ENC, ihb, hbuf, C_H, C_B, C_H, C_ENC);
    gemm(meanb, C_ENC, icw, C_ENC, icb, cbuf, C_H, C_B, C_H, C_ENC);
    h2xh_kernel<<<(C_B * C_H) / 256, 256, 0, stream>>>(hbuf, xh);

    // ---- 25 sequential decode steps ----
    for (int t = 0; t < C_S; ++t) {
        embed_kernel<<<C_B, C_E, 0, stream>>>(dec, emb, pos_emb, xh, t);
        gemm(hbuf, C_H, Ww, C_H, Wb, wah, C_ATT, C_B, C_ATT, C_H);
        attn_kernel<<<C_B, 256, 0, stream>>>(u_hs, wah, enc, Aw, Ab, alphas, xh, t);
        gemm(xh, C_XK, Wcat, C_XK, bcat, gates, C_G, C_B, C_G, C_XK);
        lstm_kernel<<<(C_B * C_H) / 256, 256, 0, stream>>>(gates, cbuf, hbuf, xh);
        gemm(hbuf, C_H, fcn_w, C_H, fcn_b, out + (size_t)t * C_V,
             (long long)C_S * C_V, C_B, C_V, C_H);
    }
}
